// QuantLlamaMLP_48137993453617
// MI455X (gfx1250) — matrix operand order repaired and checked by reading, not compiled, not run on hardware
//
#include <hip/hip_runtime.h>
#include <stdint.h>

typedef __attribute__((ext_vector_type(16))) _Float16 v16h;
typedef __attribute__((ext_vector_type(8)))  _Float16 v8h;
typedef __attribute__((ext_vector_type(8)))  float    v8f;
typedef __attribute__((ext_vector_type(4)))  int      v4i;

#define AS1 __attribute__((address_space(1)))
#define AS3 __attribute__((address_space(3)))

#if __has_builtin(__builtin_amdgcn_global_load_async_to_lds_b128)
#define HAS_ASYNC_LDS 1
#else
#define HAS_ASYNC_LDS 0
#endif

// ---- sizes ----
constexpr int D   = 4096;
constexpr int I   = 11008;
constexpr int TOK = 8192;     // 4 * 2048
constexpr int BM = 128, BN = 64, BK = 32;
constexpr int LDA = 48, LDB = 48;   // halves; 96B row stride -> 16B-aligned rows

// ---- 16-byte global -> LDS copy (async when available) ----
__device__ __forceinline__ void cp16_g2l(void* lds_dst, const void* gsrc) {
#if HAS_ASYNC_LDS
  __builtin_amdgcn_global_load_async_to_lds_b128(
      (AS1 v4i*)(uintptr_t)gsrc, (AS3 v4i*)(uintptr_t)lds_dst, 0, 0);
#else
  *(float4*)lds_dst = *(const float4*)gsrc;
#endif
}

__device__ __forceinline__ void wait_cp() {
#if HAS_ASYNC_LDS
#if __has_builtin(__builtin_amdgcn_s_wait_asynccnt)
  __builtin_amdgcn_s_wait_asynccnt(0);
#else
  asm volatile("s_wait_asynccnt 0" ::: "memory");
#endif
#endif
}

// ---- WMMA fragment loads from LDS ----
// A (16x32 f16): lane half 0 -> K 0..15 ; lane half 1 -> K 16..31
__device__ __forceinline__ v16h frag_a(const _Float16* row, int kb) {
  v8h lo = *(const v8h*)(row + kb);        // K kb..kb+7
  v8h hi = *(const v8h*)(row + kb + 8);    // K kb+8..kb+15
  v16h r;
#pragma unroll
  for (int j = 0; j < 8; ++j) { r[j] = lo[j]; r[j + 8] = hi[j]; }
  return r;
}
// B (32x16 f16): lane holds 16 contiguous K at kb = (lane<16 ? 0 : 16), col = lane&15
__device__ __forceinline__ v16h frag_b(const _Float16* p) {
  v8h lo = *(const v8h*)(p);
  v8h hi = *(const v8h*)(p + 8);
  v16h r;
#pragma unroll
  for (int j = 0; j < 8; ++j) { r[j] = lo[j]; r[j + 8] = hi[j]; }
  return r;
}

__device__ __forceinline__ v8f wmma_f16(v16h a, v16h b, v8f c) {
  return __builtin_amdgcn_wmma_f32_16x16x32_f16(false, a, false, b,
                                                (short)0, c, false, false);
}

// ================= kernel 1: x fp32 -> f16 =================
__global__ void __launch_bounds__(256)
cvt_x_kernel(const float* __restrict__ x, _Float16* __restrict__ xh) {
  size_t i = ((size_t)blockIdx.x * 256 + threadIdx.x) * 8;
  float4 a = ((const float4*)(x + i))[0];
  float4 b = ((const float4*)(x + i))[1];
  v8h o;
  o[0] = (_Float16)a.x; o[1] = (_Float16)a.y; o[2] = (_Float16)a.z; o[3] = (_Float16)a.w;
  o[4] = (_Float16)b.x; o[5] = (_Float16)b.y; o[6] = (_Float16)b.z; o[7] = (_Float16)b.w;
  *(v8h*)(xh + i) = o;
}

// ======== kernel 2: dequant + transpose  outT[n*K + k] = (q[k,n]-8)*s+z ========
__global__ void __launch_bounds__(256)
dequant_t(const int* __restrict__ q, const float* __restrict__ s,
          const float* __restrict__ z, _Float16* __restrict__ outT,
          int K, int N) {
  int n  = blockIdx.x * 256 + threadIdx.x;
  int k0 = blockIdx.y * 32;
  int g  = k0 >> 7;                       // group size 128, 32 | 128
  float sv = s[(size_t)g * N + n];
  float zv = z[(size_t)g * N + n];
  _Float16 vals[32] __attribute__((aligned(16)));
#pragma unroll
  for (int kk = 0; kk < 32; ++kk) {
    float w = ((float)q[(size_t)(k0 + kk) * N + n] - 8.0f) * sv + zv;
    vals[kk] = (_Float16)w;
  }
  _Float16* dst = outT + (size_t)n * K + k0;  // 16B aligned: K*2 % 16 == 0
#pragma unroll
  for (int j = 0; j < 4; ++j)
    ((float4*)dst)[j] = ((const float4*)vals)[j];
}

// ======== kernel 3: gate/up GEMM + fused SiLU*up -> h (f16), double-buffered ========
__global__ void __launch_bounds__(256)
gemm_gate_up(const _Float16* __restrict__ xh, const _Float16* __restrict__ WgT,
             const _Float16* __restrict__ WuT, _Float16* __restrict__ hout) {
  __shared__ _Float16 As[2][BM * LDA] __attribute__((aligned(16)));
  __shared__ _Float16 Bg[2][BN * LDB] __attribute__((aligned(16)));
  __shared__ _Float16 Bu[2][BN * LDB] __attribute__((aligned(16)));

  const int tid = threadIdx.x;
  const int lane = tid & 31, wv = tid >> 5;
  const int wm = wv & 3, wn = wv >> 2;        // 4 waves over M, 2 over N
  const int lrow = lane & 15, lhalf = lane >> 4;

  const int n0   = blockIdx.x * BN;
  const int tok0 = blockIdx.y * BM;

  // per-thread staging coordinates (fixed across iterations)
  const int am0 = (tid * 2) >> 2,       ak0 = ((tid * 2) & 3) << 3;
  const int am1 = (tid * 2 + 1) >> 2,   ak1 = ((tid * 2 + 1) & 3) << 3;
  const int bn  = tid >> 2,             bk  = (tid & 3) << 3;

  const _Float16* aSrc0 = xh  + (size_t)(tok0 + am0) * D + ak0;
  const _Float16* aSrc1 = xh  + (size_t)(tok0 + am1) * D + ak1;
  const _Float16* gSrc  = WgT + (size_t)(n0 + bn) * D + bk;
  const _Float16* uSrc  = WuT + (size_t)(n0 + bn) * D + bk;

  v8f accg[2][2], accu[2][2];
#pragma unroll
  for (int mt = 0; mt < 2; ++mt)
#pragma unroll
    for (int nt = 0; nt < 2; ++nt) { accg[mt][nt] = (v8f)0.0f; accu[mt][nt] = (v8f)0.0f; }

  constexpr int T = D / BK;

  // prologue: stage tile 0 into buffer 0
  cp16_g2l(&As[0][am0 * LDA + ak0], aSrc0);
  cp16_g2l(&As[0][am1 * LDA + ak1], aSrc1);
  cp16_g2l(&Bg[0][bn * LDB + bk], gSrc);
  cp16_g2l(&Bu[0][bn * LDB + bk], uSrc);

  for (int t = 0; t < T; ++t) {
    const int cur = t & 1;
    wait_cp();            // my tile-t copies have landed in LDS
    __syncthreads();      // everyone's have; all waves done reading buf[cur^1]

    if (t + 1 < T) {      // overlap: issue tile t+1 into the other buffer
      const int nxt = cur ^ 1;
      const int kn = (t + 1) * BK;
      cp16_g2l(&As[nxt][am0 * LDA + ak0], aSrc0 + kn);
      cp16_g2l(&As[nxt][am1 * LDA + ak1], aSrc1 + kn);
      cp16_g2l(&Bg[nxt][bn * LDB + bk], gSrc + kn);
      cp16_g2l(&Bu[nxt][bn * LDB + bk], uSrc + kn);
      if (t + 2 < T) {    // prefetch tile t+2 of the weight stream
        __builtin_prefetch((const void*)(gSrc + kn + BK), 0, 1);
        __builtin_prefetch((const void*)(uSrc + kn + BK), 0, 1);
      }
    }

    v16h af[2];
#pragma unroll
    for (int mt = 0; mt < 2; ++mt)
      af[mt] = frag_a(&As[cur][(wm * 32 + mt * 16 + lrow) * LDA], lhalf * 16);
#pragma unroll
    for (int nt = 0; nt < 2; ++nt) {
      v16h bg = frag_b(&Bg[cur][(wn * 32 + nt * 16 + lrow) * LDB + lhalf * 16]);
      v16h bu = frag_b(&Bu[cur][(wn * 32 + nt * 16 + lrow) * LDB + lhalf * 16]);
#pragma unroll
      for (int mt = 0; mt < 2; ++mt) {
        accg[mt][nt] = wmma_f16(af[mt], bg, accg[mt][nt]);
        accu[mt][nt] = wmma_f16(af[mt], bu, accu[mt][nt]);
      }
    }
  }

  // epilogue: C layout -> lanes 0-15: M=r, lanes 16-31: M=8+r; N = lane&15
#pragma unroll
  for (int mt = 0; mt < 2; ++mt)
#pragma unroll
    for (int nt = 0; nt < 2; ++nt) {
      int col  = n0 + wn * 32 + nt * 16 + lrow;
      int rowb = tok0 + wm * 32 + mt * 16 + lhalf * 8;
#pragma unroll
      for (int r = 0; r < 8; ++r) {
        float g = accg[mt][nt][r], u = accu[mt][nt][r];
        float sig = __builtin_amdgcn_rcpf(1.0f + __expf(-g));  // v_rcp_f32
        hout[(size_t)(rowb + r) * I + col] = (_Float16)(g * sig * u);
      }
    }
}

// ======== kernel 4: down GEMM: out[t,d] = h[t,:] . WdT[d,:]  (fp32 out) ========
__global__ void __launch_bounds__(256)
gemm_down(const _Float16* __restrict__ h, const _Float16* __restrict__ WdT,
          float* __restrict__ out) {
  __shared__ _Float16 As[2][BM * LDA] __attribute__((aligned(16)));
  __shared__ _Float16 Bs[2][BN * LDB] __attribute__((aligned(16)));

  const int tid = threadIdx.x;
  const int lane = tid & 31, wv = tid >> 5;
  const int wm = wv & 3, wn = wv >> 2;
  const int lrow = lane & 15, lhalf = lane >> 4;

  const int n0   = blockIdx.x * BN;
  const int tok0 = blockIdx.y * BM;

  const int am0 = (tid * 2) >> 2,       ak0 = ((tid * 2) & 3) << 3;
  const int am1 = (tid * 2 + 1) >> 2,   ak1 = ((tid * 2 + 1) & 3) << 3;
  const int bn  = tid >> 2,             bk  = (tid & 3) << 3;

  const _Float16* aSrc0 = h   + (size_t)(tok0 + am0) * I + ak0;
  const _Float16* aSrc1 = h   + (size_t)(tok0 + am1) * I + ak1;
  const _Float16* bSrc  = WdT + (size_t)(n0 + bn) * I + bk;

  v8f acc[2][2];
#pragma unroll
  for (int mt = 0; mt < 2; ++mt)
#pragma unroll
    for (int nt = 0; nt < 2; ++nt) acc[mt][nt] = (v8f)0.0f;

  constexpr int T = I / BK;

  cp16_g2l(&As[0][am0 * LDA + ak0], aSrc0);
  cp16_g2l(&As[0][am1 * LDA + ak1], aSrc1);
  cp16_g2l(&Bs[0][bn * LDB + bk], bSrc);

  for (int t = 0; t < T; ++t) {
    const int cur = t & 1;
    wait_cp();
    __syncthreads();

    if (t + 1 < T) {
      const int nxt = cur ^ 1;
      const int kn = (t + 1) * BK;
      cp16_g2l(&As[nxt][am0 * LDA + ak0], aSrc0 + kn);
      cp16_g2l(&As[nxt][am1 * LDA + ak1], aSrc1 + kn);
      cp16_g2l(&Bs[nxt][bn * LDB + bk], bSrc + kn);
      if (t + 2 < T)
        __builtin_prefetch((const void*)(bSrc + kn + BK), 0, 1);
    }

    v16h af[2];
#pragma unroll
    for (int mt = 0; mt < 2; ++mt)
      af[mt] = frag_a(&As[cur][(wm * 32 + mt * 16 + lrow) * LDA], lhalf * 16);
#pragma unroll
    for (int nt = 0; nt < 2; ++nt) {
      v16h bf = frag_b(&Bs[cur][(wn * 32 + nt * 16 + lrow) * LDB + lhalf * 16]);
#pragma unroll
      for (int mt = 0; mt < 2; ++mt)
        acc[mt][nt] = wmma_f16(af[mt], bf, acc[mt][nt]);
    }
  }

#pragma unroll
  for (int mt = 0; mt < 2; ++mt)
#pragma unroll
    for (int nt = 0; nt < 2; ++nt) {
      int col  = n0 + wn * 32 + nt * 16 + lrow;
      int rowb = tok0 + wm * 32 + mt * 16 + lhalf * 8;
#pragma unroll
      for (int r = 0; r < 8; ++r)
        out[(size_t)(rowb + r) * D + col] = acc[mt][nt][r];
    }
}

// ================= host launcher =================
extern "C" void kernel_launch(void* const* d_in, const int* in_sizes, int n_in,
                              void* d_out, int out_size, void* d_ws, size_t ws_size,
                              hipStream_t stream) {
  const float* x  = (const float*)d_in[0];
  const int*   gq = (const int*)d_in[1];
  const float* gs = (const float*)d_in[2];
  const float* gz = (const float*)d_in[3];
  const int*   uq = (const int*)d_in[4];
  const float* us = (const float*)d_in[5];
  const float* uz = (const float*)d_in[6];
  const int*   dq = (const int*)d_in[7];
  const float* dsc = (const float*)d_in[8];
  const float* dz = (const float*)d_in[9];
  float* out = (float*)d_out;
  char* ws = (char*)d_ws;

  // workspace layout (bytes)
  _Float16* xh  = (_Float16*)(ws);                    // 8192*4096*2   = 67,108,864
  _Float16* WgT = (_Float16*)(ws + 67108864ull);      // 11008*4096*2  = 90,177,536
  _Float16* WuT = (_Float16*)(ws + 157286400ull);
  _Float16* WdT = (_Float16*)(ws + 247463936ull);     // 4096 rows x 11008 k
  _Float16* hbuf= (_Float16*)(ws + 337641472ull);     // 8192*11008*2  = 180,355,072

  cvt_x_kernel<<<dim3((TOK * D) / (256 * 8)), 256, 0, stream>>>(x, xh);
  dequant_t<<<dim3(I / 256, D / 32), 256, 0, stream>>>(gq, gs, gz, WgT, D, I);
  dequant_t<<<dim3(I / 256, D / 32), 256, 0, stream>>>(uq, us, uz, WuT, D, I);
  dequant_t<<<dim3(D / 256, I / 32), 256, 0, stream>>>(dq, dsc, dz, WdT, I, D);
  gemm_gate_up<<<dim3(I / BN, TOK / BM), 256, 0, stream>>>(xh, WgT, WuT, hbuf);
  gemm_down<<<dim3(D / BN, TOK / BM), 256, 0, stream>>>(hbuf, WdT, out);
}
